// MultiHeadAttention_64725157151059
// MI455X (gfx1250) — matrix operand order repaired and checked by reading, not compiled, not run on hardware
//
#include <hip/hip_runtime.h>
#include <hip/hip_bf16.h>
#include <stdint.h>

typedef __bf16 bf16_t;
typedef __bf16 v8bf  __attribute__((ext_vector_type(8)));
typedef __bf16 v16bf __attribute__((ext_vector_type(16)));
typedef float  v8f   __attribute__((ext_vector_type(8)));

#define D_MODEL 512
#define N_HEADS 8
#define D_K     64
#define SEQ     4096
#define BATCH   2
#define MTOT    (BATCH*SEQ)          // 8192 rows total

#define PST 72   // P restage LDS pitch for 16x64 tile: 144 B/row (16-B aligned)

// softmax scale folded into Q: 1/sqrt(64) * log2(e)
#define QSCALE 0.1803368801111244f
#define NEG_BIG (-1.0e30f)

static __device__ __forceinline__ v16bf cat8(v8bf lo, v8bf hi) {
  return __builtin_shufflevector(lo, hi, 0,1,2,3,4,5,6,7,8,9,10,11,12,13,14,15);
}

static __device__ __forceinline__ v16bf ld_frag(const bf16_t* p) {
  // bf16 p[0..7] and p[16..23] (16-B aligned) -> one fragment
  return cat8(*(const v8bf*)p, *(const v8bf*)(p + 16));
}

static __device__ __forceinline__ v8f wmma_bf16(v16bf a, v16bf b, v8f c) {
  return __builtin_amdgcn_wmma_f32_16x16x32_bf16(false, a, false, b, (short)0, c, false, false);
}

// ---------------------------------------------------------------- convert ---
__global__ void cvt_f32_bf16(const float* __restrict__ src,
                             bf16_t* __restrict__ dst, int n) {
  int i = blockIdx.x * blockDim.x + threadIdx.x;
  if (i < n) dst[i] = (bf16_t)src[i];
}

// src is KxN row-major; dst is NxK row-major (dst[n][k] = src[k][n])
__global__ void cvt_transpose_f32_bf16(const float* __restrict__ src,
                                       bf16_t* __restrict__ dst, int K, int N) {
  int i = blockIdx.x * blockDim.x + threadIdx.x;
  if (i < N * K) {
    int n = i / K, k = i % K;
    dst[i] = (bf16_t)src[(size_t)k * N + n];
  }
}

// ------------------------------------------------------ GEMM core (2m x 4n) -
// Wave computes a 32x64 C tile. A row-major [M][512] bf16, Bt pre-transposed
// [N][512] bf16. K-loop fully unrolled: every fragment is two b128 loads with
// an immediate offset off 6 loop-invariant pointers. 12 loads per 8 wmma.
struct Acc24 { v8f a[2][4]; };

static __device__ __forceinline__ Acc24
gemm_wave_32x64(const bf16_t* __restrict__ A, const bf16_t* __restrict__ Bt,
                int mbase, int nbase) {
  const int lane = threadIdx.x & 31;
  const int half = lane >> 4, ln16 = lane & 15;
  Acc24 acc;
  #pragma unroll
  for (int mi = 0; mi < 2; ++mi)
    #pragma unroll
    for (int ni = 0; ni < 4; ++ni)
      #pragma unroll
      for (int r = 0; r < 8; ++r) acc.a[mi][ni][r] = 0.f;

  const bf16_t* arow[2];
  const bf16_t* brow[4];
  #pragma unroll
  for (int mi = 0; mi < 2; ++mi)
    arow[mi] = A  + (size_t)(mbase + mi*16 + ln16) * D_MODEL + half*8;
  #pragma unroll
  for (int ni = 0; ni < 4; ++ni)
    brow[ni] = Bt + (size_t)(nbase + ni*16 + ln16) * D_MODEL + half*8;

  #pragma unroll
  for (int k0 = 0; k0 < D_MODEL; k0 += 32) {
    v16bf a[2], b[4];
    #pragma unroll
    for (int mi = 0; mi < 2; ++mi)   // A frag: elems {half*8+0..7, 16+half*8+0..7}
      a[mi] = cat8(*(const v8bf*)(arow[mi] + k0),
                   *(const v8bf*)(arow[mi] + k0 + 16));
    #pragma unroll
    for (int ni = 0; ni < 4; ++ni)   // B frag: elems {half*8+0..7, 16+half*8+0..7}
      b[ni] = ld_frag(brow[ni] + k0);
    #pragma unroll
    for (int mi = 0; mi < 2; ++mi)
      #pragma unroll
      for (int ni = 0; ni < 4; ++ni)
        acc.a[mi][ni] = wmma_bf16(a[mi], b[ni], acc.a[mi][ni]);
  }
  return acc;
}

// -------------------------------------------------------------- QKV GEMM ---
// cols [0,512)=Q (pre-scaled by QSCALE) -> Qb[bh][t][dk],
// [512,1024)=K -> Kb[bh][t][dk], [1024,1536)=V -> Vtb[bh][dk][t].
__global__ void gemm_qkv(const bf16_t* __restrict__ xb,
                         const bf16_t* __restrict__ Wt,   // [1536][512]
                         const float*  __restrict__ bias,
                         bf16_t* __restrict__ Qb, bf16_t* __restrict__ Kb,
                         bf16_t* __restrict__ Vtb) {
  const int wave = threadIdx.x >> 5;
  const int lane = threadIdx.x & 31;
  const int half = lane >> 4, ln16 = lane & 15;
  const int nbase = (blockIdx.x * 8 + wave) * 64;   // 64-aligned -> h uniform
  const int mbase = blockIdx.y * 32;

  Acc24 acc = gemm_wave_32x64(xb, Wt, mbase, nbase);

  const int which = nbase / D_MODEL;                // 0=Q 1=K 2=V, uniform
  const int h     = (nbase % D_MODEL) / D_K;        // uniform
  #pragma unroll
  for (int ni = 0; ni < 4; ++ni) {
    const int ng = nbase + ni*16 + ln16;
    const float bv = bias[ng];
    const int dk = ni*16 + ln16;                    // nbase is 64-aligned
    #pragma unroll
    for (int mi = 0; mi < 2; ++mi) {
      #pragma unroll
      for (int r = 0; r < 8; ++r) {
        int mg = mbase + mi*16 + r + 8*half;
        int b  = mg >> 12, t = mg & (SEQ - 1);
        size_t bh = (size_t)(b*N_HEADS + h);
        float v = acc.a[mi][ni][r] + bv;
        if (which == 0)       Qb [(bh*SEQ + t)*D_K + dk] = (bf16_t)(v * QSCALE);
        else if (which == 1)  Kb [(bh*SEQ + t)*D_K + dk] = (bf16_t)v;
        else                  Vtb[(bh*D_K + dk)*SEQ + t] = (bf16_t)v;
      }
    }
  }
}

// -------------------------------------------------------- flash attention ---
// 256 threads = 8 independent waves; each wave owns one 16-row q tile and
// streams 64-wide kv blocks via two incrementing per-lane pointers; all tile
// displacements are instruction-immediate offsets. Only the diagonal block
// is masked. Softmax runs in the log2 domain (scale folded into Q).
__global__ void flash_attn(const bf16_t* __restrict__ Qb,
                           const bf16_t* __restrict__ Kb,
                           const bf16_t* __restrict__ Vtb,
                           bf16_t* __restrict__ attn) {
  __shared__ __align__(16) bf16_t Pst[8*16*PST];

  const int lane = threadIdx.x & 31, wave = threadIdx.x >> 5;
  const int half = lane >> 4,       ln16 = lane & 15;
  const int qt = blockIdx.x * 8 + wave;        // 16-row q tile, 0..255
  const int bh = blockIdx.y;                   // 0..15
  const int b  = bh >> 3, h = bh & 7;
  const size_t headTD = (size_t)bh * SEQ * D_K;   // Q/K [t][dk] base
  const size_t headDT = (size_t)bh * D_K * SEQ;   // V^T [dk][t] base

  // Q fragments: 2 k-steps of 32 over dk=64 (Q already carries QSCALE)
  v16bf qf[2];
  {
    const bf16_t* qrow = Qb + headTD + (size_t)(qt*16 + ln16)*D_K;
    #pragma unroll
    for (int kk = 0; kk < 2; ++kk)
      qf[kk] = cat8(*(const v8bf*)&qrow[kk*32 + half*8],
                    *(const v8bf*)&qrow[kk*32 + 16 + half*8]);
  }

  v8f O[4];
  #pragma unroll
  for (int nt = 0; nt < 4; ++nt)
    #pragma unroll
    for (int r = 0; r < 8; ++r) O[nt][r] = 0.f;
  float mrun[8], lrun[8];
  #pragma unroll
  for (int r = 0; r < 8; ++r) { mrun[r] = NEG_BIG; lrun[r] = 0.f; }

  bf16_t* pw = &Pst[wave*16*PST];
  const bf16_t* pr = &Pst[wave*16*PST + ln16*PST + half*8];  // A-frag read base
  const int jmax = (qt >> 2) + 1;              // 64-wide causal bound

  // loop-invariant per-lane operand pointers; advance by constants per j
  const bf16_t* kl = Kb  + headTD + (size_t)ln16*D_K + half*8;  // +s*1024+kk*32
  const bf16_t* vl = Vtb + headDT + (size_t)ln16*SEQ + half*8;  // +nt*16*SEQ+kk*32
  const char*   kpf = (const char*)(Kb + headTD + (size_t)64*D_K) + lane*256;
  const char*   vpf = (const char*)(Vtb + headDT + 64) + lane*256;

  for (int j = 0; j < jmax; ++j) {
    // prefetch next kv block (64 rows x 128 B = 64 lines)
    if (j + 1 < jmax) {
      __builtin_prefetch(kpf, 0, 3);
      __builtin_prefetch(kpf + 128, 0, 3);
      __builtin_prefetch(vpf, 0, 3);
    }

    // ---- scores: Q(16x64) @ K^T(64x64) -> four 16x16 subtiles
    v8f sc[4];
    #pragma unroll
    for (int s = 0; s < 4; ++s) {
      v8f c0 = {0.f,0.f,0.f,0.f,0.f,0.f,0.f,0.f};
      #pragma unroll
      for (int kk = 0; kk < 2; ++kk)
        c0 = wmma_bf16(qf[kk], ld_frag(kl + s*(16*D_K) + kk*32), c0);
      sc[s] = c0;
    }

    // ---- causal mask: only the diagonal block needs it (wave-uniform)
    if (j == jmax - 1) {
      const int qrow0 = qt*16 + 8*half;        // + r
      const int col0  = j*64 + ln16;           // + s*16
      #pragma unroll
      for (int s = 0; s < 4; ++s)
        #pragma unroll
        for (int r = 0; r < 8; ++r)
          if (col0 + s*16 > qrow0 + r) sc[s][r] = NEG_BIG;
    }

    // ---- online softmax in log2 domain (reduce across 16-lane half)
    #pragma unroll
    for (int r = 0; r < 8; ++r) {
      float rm = fmaxf(fmaxf(sc[0][r], sc[1][r]), fmaxf(sc[2][r], sc[3][r]));
      rm = fmaxf(rm, __shfl_xor(rm, 1, 32));
      rm = fmaxf(rm, __shfl_xor(rm, 2, 32));
      rm = fmaxf(rm, __shfl_xor(rm, 4, 32));
      rm = fmaxf(rm, __shfl_xor(rm, 8, 32));
      float mnew = fmaxf(mrun[r], rm);
      float rs = 0.f;
      #pragma unroll
      for (int s = 0; s < 4; ++s) {
        float p = __builtin_exp2f(sc[s][r] - mnew);
        sc[s][r] = p;
        rs += p;
      }
      rs += __shfl_xor(rs, 1, 32);
      rs += __shfl_xor(rs, 2, 32);
      rs += __shfl_xor(rs, 4, 32);
      rs += __shfl_xor(rs, 8, 32);
      float alpha = __builtin_exp2f(mrun[r] - mnew);
      lrun[r] = lrun[r]*alpha + rs;
      mrun[r] = mnew;
      #pragma unroll
      for (int nt = 0; nt < 4; ++nt) O[nt][r] *= alpha;
    }

    // ---- restage P 16x64 (C layout -> A fragment layout) via per-wave LDS
    #pragma unroll
    for (int s = 0; s < 4; ++s)
      #pragma unroll
      for (int r = 0; r < 8; ++r)
        pw[(r + 8*half)*PST + s*16 + ln16] = (bf16_t)sc[s][r];
    asm volatile("s_wait_dscnt 0" ::: "memory");   // cross-lane LDS RAW
    v16bf pf[2];
    #pragma unroll
    for (int kk = 0; kk < 2; ++kk)
      pf[kk] = cat8(*(const v8bf*)(pr + kk*32),
                    *(const v8bf*)(pr + kk*32 + 16));

    // ---- O += P(16x64) @ V(64x64): 2 k-steps x 4 n-tiles, V^T contiguous
    #pragma unroll
    for (int kk = 0; kk < 2; ++kk)
      #pragma unroll
      for (int nt = 0; nt < 4; ++nt)
        O[nt] = wmma_bf16(pf[kk], ld_frag(vl + nt*(16*SEQ) + kk*32), O[nt]);

    kl  += 64*D_K;
    vl  += 64;
    kpf += 64*D_K*sizeof(bf16_t);
    vpf += 64*sizeof(bf16_t);
  }

  // ---- normalize and write attn (bf16, [B][T][D], col = h*64 + dk)
  #pragma unroll
  for (int r = 0; r < 8; ++r) {
    float inv = 1.0f / lrun[r];
    int t = qt*16 + r + 8*half;
    size_t rowo = ((size_t)b*SEQ + t)*D_MODEL + h*D_K;
    #pragma unroll
    for (int nt = 0; nt < 4; ++nt)
      attn[rowo + nt*16 + ln16] = (bf16_t)(O[nt][r] * inv);
  }
}

// ------------------------------------------------------------- out GEMM ----
__global__ void gemm_out(const bf16_t* __restrict__ ab,
                         const bf16_t* __restrict__ Wt,   // [512][512]
                         const float*  __restrict__ bias,
                         float* __restrict__ out) {
  const int wave = threadIdx.x >> 5;
  const int lane = threadIdx.x & 31;
  const int half = lane >> 4, ln16 = lane & 15;
  const int nbase = (blockIdx.x * 8 + wave) * 64;
  const int mbase = blockIdx.y * 32;

  Acc24 acc = gemm_wave_32x64(ab, Wt, mbase, nbase);

  #pragma unroll
  for (int ni = 0; ni < 4; ++ni) {
    const int ng = nbase + ni*16 + ln16;
    const float bv = bias[ng];
    #pragma unroll
    for (int mi = 0; mi < 2; ++mi)
      #pragma unroll
      for (int r = 0; r < 8; ++r) {
        int mg = mbase + mi*16 + r + 8*half;
        out[(size_t)mg*D_MODEL + ng] = acc.a[mi][ni][r] + bv;
      }
  }
}

// ---------------------------------------------------------------- launch ---
extern "C" void kernel_launch(void* const* d_in, const int* in_sizes, int n_in,
                              void* d_out, int out_size, void* d_ws, size_t ws_size,
                              hipStream_t stream) {
  const float* x     = (const float*)d_in[0];
  const float* W_qkv = (const float*)d_in[1];
  const float* b_qkv = (const float*)d_in[2];
  const float* W_out = (const float*)d_in[3];
  const float* b_out = (const float*)d_in[4];
  float* out = (float*)d_out;

  char* ws = (char*)d_ws;
  size_t off = 0;
  auto take = [&](size_t bytes) -> char* {
    char* p = ws + off;
    off += (bytes + 255) & ~(size_t)255;
    return p;
  };
  bf16_t* xb   = (bf16_t*)take((size_t)MTOT*D_MODEL*sizeof(bf16_t));
  bf16_t* Wqt  = (bf16_t*)take((size_t)3*D_MODEL*D_MODEL*sizeof(bf16_t)); // [1536][512]
  bf16_t* Wot  = (bf16_t*)take((size_t)D_MODEL*D_MODEL*sizeof(bf16_t));   // [512][512]
  bf16_t* Qb   = (bf16_t*)take((size_t)MTOT*D_MODEL*sizeof(bf16_t));
  bf16_t* Kb   = (bf16_t*)take((size_t)MTOT*D_MODEL*sizeof(bf16_t));
  bf16_t* Vtb  = (bf16_t*)take((size_t)MTOT*D_MODEL*sizeof(bf16_t));
  bf16_t* ab   = (bf16_t*)take((size_t)MTOT*D_MODEL*sizeof(bf16_t));

  {
    int n = MTOT*D_MODEL;
    cvt_f32_bf16<<<(n + 255)/256, 256, 0, stream>>>(x, xb, n);
  }
  {
    int n = 3*D_MODEL*D_MODEL;   // W_qkv: 512 x 1536 -> Wqt: 1536 x 512
    cvt_transpose_f32_bf16<<<(n + 255)/256, 256, 0, stream>>>(
        W_qkv, Wqt, D_MODEL, 3*D_MODEL);
  }
  {
    int n = D_MODEL*D_MODEL;     // W_out: 512 x 512 -> Wot
    cvt_transpose_f32_bf16<<<(n + 255)/256, 256, 0, stream>>>(
        W_out, Wot, D_MODEL, D_MODEL);
  }

  // QKV: M=8192, N=1536; per block: 8 waves x 64 cols = 512 cols
  gemm_qkv<<<dim3(3*D_MODEL/512, MTOT/32), 256, 0, stream>>>(
      xb, Wqt, b_qkv, Qb, Kb, Vtb);

  // attention: 8 q-tiles per block, one per wave
  flash_attn<<<dim3(SEQ/128, BATCH*N_HEADS), 256, 0, stream>>>(
      Qb, Kb, Vtb, ab);

  // out proj: M=8192, N=512; 8 waves x 64 cols = 512 cols per block
  gemm_out<<<dim3(1, MTOT/32), 256, 0, stream>>>(
      ab, Wot, b_out, out);
}
